// RelativeMultiHeadAttn_46136538693893
// MI455X (gfx1250) — matrix operand order repaired and checked by reading, not compiled, not run on hardware
//
#include <hip/hip_runtime.h>

// ---------------------------------------------------------------------------
// Transformer-XL relative multi-head attention block for MI455X (gfx1250).
// GEMMs on v_wmma_f32_16x16x32_bf16; large GEMMs stage tiles into LDS with
// gfx1250 async copies (global_load_async_to_lds_b128 + s_wait_asynccnt),
// double-buffered so the copy of stage i+1 overlaps the WMMAs of stage i.
// ---------------------------------------------------------------------------

constexpr int BATCH = 2;
constexpr int SEQ   = 2048;
constexpr int DM    = 1024;   // d_model
constexpr int NH    = 16;     // heads
constexpr int DH    = 64;     // head dim
constexpr int N3    = 3 * DM; // qkv width
constexpr int BS    = BATCH * SEQ;

typedef __attribute__((ext_vector_type(16))) __bf16 v16bf;
typedef __attribute__((ext_vector_type(8)))  __bf16 v8bf;
typedef __attribute__((ext_vector_type(8)))  float  v8f;

__device__ __forceinline__ v8f wmma_bf16(v16bf a, v16bf b, v8f c) {
  // (neg_a, A, neg_b, B, c_mod, C, reuse_a, reuse_b)
  return __builtin_amdgcn_wmma_f32_16x16x32_bf16(false, a, false, b, (short)0, c,
                                                 false, false);
}

// Async DMA: 16B per lane, global -> LDS, tracked by ASYNCcnt (no VGPR data).
__device__ __forceinline__ void async_b128(unsigned lds_off, const __bf16* gp) {
  unsigned long long ga = (unsigned long long)(uintptr_t)gp;
  asm volatile("global_load_async_to_lds_b128 %0, %1, off"
               :: "v"(lds_off), "v"(ga) : "memory");
}

// ---------------------------------------------------------------------------
// WMMA fragment helpers.
// A fragment (16x32 bf16, row-major, leading dim ld):
//   lane: row = base+(lane&15); e<8 -> K = kb+hf*8+e ; e>=8 -> K = kb+16+hf*8+(e-8)
// B fragment (32x16) read from TRANSPOSED [N,K] storage:
//   lane: col = base+(lane&15); e<8 -> K = kb+hf*8+e ; e>=8 -> K = kb+16+hf*8+(e-8)
// ---------------------------------------------------------------------------
__device__ __forceinline__ v16bf load_a_frag(const __bf16* src, int row, int ld,
                                             int kb, int lane) {
  int hf = lane >> 4;
  const __bf16* p = src + (size_t)row * ld + kb + hf * 8;
  v8bf lo = *(const v8bf*)p;
  v8bf hi = *(const v8bf*)(p + 16);
  v16bf r;
#pragma unroll
  for (int i = 0; i < 8; ++i) { r[i] = lo[i]; r[i + 8] = hi[i]; }
  return r;
}

__device__ __forceinline__ v16bf load_b_fragT(const __bf16* srcT, int col, int ld,
                                              int kb, int lane) {
  int hf = lane >> 4;
  const __bf16* p = srcT + (size_t)col * ld + kb + hf * 8;
  v8bf lo = *(const v8bf*)p;
  v8bf hi = *(const v8bf*)(p + 16);
  v16bf r;
#pragma unroll
  for (int i = 0; i < 8; ++i) { r[i] = lo[i]; r[i + 8] = hi[i]; }
  return r;
}

// LDS-tile fragment readers; tile rows padded to TPAD halves (80B) for banks.
constexpr int TPAD = 40;

__device__ __forceinline__ v16bf lds_a_frag(const __bf16* tile, int l16, int hf) {
  const __bf16* p = tile + l16 * TPAD + hf * 8;
  v8bf lo = *(const v8bf*)p;
  v8bf hi = *(const v8bf*)(p + 16);
  v16bf r;
#pragma unroll
  for (int i = 0; i < 8; ++i) { r[i] = lo[i]; r[i + 8] = hi[i]; }
  return r;
}

__device__ __forceinline__ v16bf lds_b_frag(const __bf16* tile, int l16, int hf) {
  const __bf16* p = tile + l16 * TPAD + hf * 8;
  v8bf lo = *(const v8bf*)p;
  v8bf hi = *(const v8bf*)(p + 16);
  v16bf r;
#pragma unroll
  for (int i = 0; i < 8; ++i) { r[i] = lo[i]; r[i + 8] = hi[i]; }
  return r;
}

// ---------------------------------------------------------------------------
// Double-buffered LDS-staged 64x128 workgroup GEMM tile over K = DM.
// WG = 8 waves (2 M-groups x 4 N-groups), wave tile 32x32 (4 accumulators).
// Per stage each wave issues exactly 3 async b128 ops (A:1, B:2) -> wait 0x3
// retires the previous stage while the next streams in.
// ---------------------------------------------------------------------------
__device__ __forceinline__ void gemm_tile_staged(const __bf16* __restrict__ Ag,
                                                 const __bf16* __restrict__ Bg,
                                                 int m0_wg, int n0_wg,
                                                 v8f acc[2][2]) {
  __shared__ __align__(16) __bf16 Ash[2][64 * TPAD];
  __shared__ __align__(16) __bf16 Bsh[2][128 * TPAD];
  const int tid  = threadIdx.x;
  const int lane = tid & 31, wave = tid >> 5;
  const int l16 = lane & 15, hf = lane >> 4;
  const int r = tid >> 2, seg = tid & 3;  // 64 rows x 4 16B segments

  auto stage = [&](int buf, int k0) {
    async_b128((unsigned)(uintptr_t)&Ash[buf][r * TPAD + seg * 8],
               Ag + (size_t)(m0_wg + r) * DM + k0 + seg * 8);
    async_b128((unsigned)(uintptr_t)&Bsh[buf][r * TPAD + seg * 8],
               Bg + (size_t)(n0_wg + r) * DM + k0 + seg * 8);
    async_b128((unsigned)(uintptr_t)&Bsh[buf][(r + 64) * TPAD + seg * 8],
               Bg + (size_t)(n0_wg + r + 64) * DM + k0 + seg * 8);
  };

  stage(0, 0);
  int buf = 0;
  for (int k0 = 0; k0 < DM; k0 += 32) {
    const bool more = (k0 + 32 < DM);
    if (more) {
      stage(buf ^ 1, k0 + 32);
      asm volatile("s_wait_asynccnt 0x3" ::: "memory");  // previous stage landed
    } else {
      asm volatile("s_wait_asynccnt 0x0" ::: "memory");
    }
    __syncthreads();
    const __bf16* At = &Ash[buf][(wave >> 2) * 32 * TPAD];
    const __bf16* Bt = &Bsh[buf][(wave & 3) * 32 * TPAD];
    v16bf a0 = lds_a_frag(At, l16, hf);
    v16bf a1 = lds_a_frag(At + 16 * TPAD, l16, hf);
    v16bf b0 = lds_b_frag(Bt, l16, hf);
    v16bf b1 = lds_b_frag(Bt + 16 * TPAD, l16, hf);
    acc[0][0] = wmma_bf16(a0, b0, acc[0][0]);
    acc[0][1] = wmma_bf16(a0, b1, acc[0][1]);
    acc[1][0] = wmma_bf16(a1, b0, acc[1][0]);
    acc[1][1] = wmma_bf16(a1, b1, acc[1][1]);
    __syncthreads();  // everyone done reading buf before it is restaged
    buf ^= 1;
  }
}

// ---------------------------------------------------------------------------
// fp32 -> bf16 converters
// ---------------------------------------------------------------------------
__global__ __launch_bounds__(256) void k_cvt(const float* s, __bf16* d, int n) {
  int i = blockIdx.x * 256 + threadIdx.x;
  if (i < n) d[i] = (__bf16)s[i];
}

// s[K][N] (row-major) -> d[N][K]
__global__ __launch_bounds__(256) void k_cvtT(const float* s, __bf16* d, int K, int N) {
  int i = blockIdx.x * 256 + threadIdx.x;
  if (i < K * N) {
    int k = i / N, n = i - k * N;
    d[(size_t)n * K + k] = (__bf16)s[i];
  }
}

// ---------------------------------------------------------------------------
// GEMM1: qkv = x @ w_qkv + b_qkv ; scatter to q (scaled by 1/8), k, v^T
// ---------------------------------------------------------------------------
__global__ __launch_bounds__(256) void k_gemm_qkv(const __bf16* __restrict__ xb,
                                                  const __bf16* __restrict__ wT,
                                                  const float* __restrict__ bias,
                                                  __bf16* __restrict__ qb,
                                                  __bf16* __restrict__ kb,
                                                  __bf16* __restrict__ vT) {
  int lane = threadIdx.x & 31, wave = threadIdx.x >> 5;
  int l16 = lane & 15, hf = lane >> 4;
  int m0 = blockIdx.y * 64 + (wave >> 2) * 32;
  int n0 = blockIdx.x * 128 + (wave & 3) * 32;
  v8f acc[2][2] = {};
  gemm_tile_staged(xb, wT, blockIdx.y * 64, blockIdx.x * 128, acc);
#pragma unroll
  for (int i = 0; i < 2; ++i)
#pragma unroll
    for (int j = 0; j < 2; ++j) {
      int n = n0 + 16 * j + l16;
      float bv = bias[n];
      int which = n >> 10, nn = n & 1023;
      int h = nn >> 6, d = nn & 63;
#pragma unroll
      for (int v = 0; v < 8; ++v) {
        int m = m0 + 16 * i + v + 8 * hf;
        int bI = m >> 11, s = m & (SEQ - 1);
        float val = acc[i][j][v] + bv;
        size_t head = (size_t)(bI * NH + h);
        if (which == 0)       // q, fold in 1/sqrt(DH)=0.125 (covers ac AND bd)
          qb[(head * SEQ + s) * DH + d] = (__bf16)(val * 0.125f);
        else if (which == 1)  // k
          kb[(head * SEQ + s) * DH + d] = (__bf16)val;
        else                  // v, stored transposed [dh, S]
          vT[(head * DH + d) * SEQ + s] = (__bf16)val;
      }
    }
}

// ---------------------------------------------------------------------------
// bd_raw[SEQ,SEQ] = q_h @ p_h^T for one (b,h).  K = DH = 64 (2 steps) ->
// staging cannot pay off; direct-from-L2 fragments.
// ---------------------------------------------------------------------------
__global__ __launch_bounds__(256) void k_gemm_bd(const __bf16* __restrict__ qb,
                                                 const __bf16* __restrict__ posb,
                                                 float* __restrict__ bdraw,
                                                 int b, int h) {
  int lane = threadIdx.x & 31, wave = threadIdx.x >> 5;
  int l16 = lane & 15, hf = lane >> 4;
  int m0 = blockIdx.y * 64 + (wave >> 2) * 32;
  int n0 = blockIdx.x * 128 + (wave & 3) * 32;
  const __bf16* A = qb + (size_t)(b * NH + h) * SEQ * DH;
  const __bf16* P = posb + h * DH;  // leading dim DM
  v8f acc[2][2] = {};
#pragma unroll
  for (int k0 = 0; k0 < DH; k0 += 32) {
    v16bf a0 = load_a_frag(A, m0 + l16, DH, k0, lane);
    v16bf a1 = load_a_frag(A, m0 + 16 + l16, DH, k0, lane);
    v16bf b0 = load_b_fragT(P, n0 + l16, DM, k0, lane);
    v16bf b1 = load_b_fragT(P, n0 + 16 + l16, DM, k0, lane);
    acc[0][0] = wmma_bf16(a0, b0, acc[0][0]);
    acc[0][1] = wmma_bf16(a0, b1, acc[0][1]);
    acc[1][0] = wmma_bf16(a1, b0, acc[1][0]);
    acc[1][1] = wmma_bf16(a1, b1, acc[1][1]);
  }
#pragma unroll
  for (int i = 0; i < 2; ++i)
#pragma unroll
    for (int j = 0; j < 2; ++j)
#pragma unroll
      for (int v = 0; v < 8; ++v) {
        int m = m0 + 16 * i + v + 8 * hf;
        int n = n0 + 16 * j + l16;
        bdraw[(size_t)m * SEQ + n] = acc[i][j][v];
      }
}

// ---------------------------------------------------------------------------
// Fused attention for one (b,h): flash-style online softmax.
// Each wave owns 16 query rows; keys processed 32 at a time.
// scores = q@k^T (WMMA) + rel-shift gather from bd_raw (exact permutation:
//   f = S + q*S + k; qi = f/(S+1); j = f%(S+1); val = j ? raw[qi][j-1] : 0 )
// P converted C-layout -> A-fragment via LDS, then O += P @ V^T (WMMA).
// ---------------------------------------------------------------------------
constexpr int PST = 40;  // padded LDS row stride (bf16 elems)

__global__ __launch_bounds__(256) void k_attn(const __bf16* __restrict__ qb,
                                              const __bf16* __restrict__ kb,
                                              const __bf16* __restrict__ vT,
                                              const float* __restrict__ bdraw,
                                              __bf16* __restrict__ attnout,
                                              int b, int h) {
  __shared__ __align__(32) __bf16 pshm[8][16 * PST];
  int lane = threadIdx.x & 31, wave = threadIdx.x >> 5;
  int l16 = lane & 15, hf = lane >> 4;
  int q0 = blockIdx.x * 128 + wave * 16;
  size_t head = (size_t)(b * NH + h);
  const __bf16* Q = qb + head * SEQ * DH;
  const __bf16* Kt = kb + head * SEQ * DH;  // row-major [S,dh] == B^T layout
  const __bf16* Vt = vT + head * DH * SEQ;  // [dh, S]
  __bf16* pbuf = &pshm[wave][0];

  v16bf aq0 = load_a_frag(Q, q0 + l16, DH, 0, lane);
  v16bf aq1 = load_a_frag(Q, q0 + l16, DH, 32, lane);

  float mrun[8], lrun[8];
  v8f oacc[4] = {};
#pragma unroll
  for (int v = 0; v < 8; ++v) { mrun[v] = -1e30f; lrun[v] = 0.f; }

  for (int kt = 0; kt < SEQ; kt += 32) {
    // ---- scores: two 16x16 tiles over 32 key columns ---------------------
    v8f s0 = {}, s1 = {};
    {
      v16bf b0 = load_b_fragT(Kt, kt + l16, DH, 0, lane);
      v16bf b1 = load_b_fragT(Kt, kt + l16, DH, 32, lane);
      s0 = wmma_bf16(aq0, b0, s0);
      s0 = wmma_bf16(aq1, b1, s0);
      b0 = load_b_fragT(Kt, kt + 16 + l16, DH, 0, lane);
      b1 = load_b_fragT(Kt, kt + 16 + l16, DH, 32, lane);
      s1 = wmma_bf16(aq0, b0, s1);
      s1 = wmma_bf16(aq1, b1, s1);
    }
    // ---- rel-shift gather + online softmax update ------------------------
    float p0[8], p1[8];
#pragma unroll
    for (int v = 0; v < 8; ++v) {
      int qrow = q0 + v + 8 * hf;
      unsigned f0 = (unsigned)(SEQ + qrow * SEQ + kt + l16);
      unsigned qi = f0 / (unsigned)(SEQ + 1);
      unsigned jr = f0 - qi * (unsigned)(SEQ + 1);
      float bd0 = jr ? bdraw[(size_t)qi * SEQ + jr - 1] : 0.f;
      unsigned f1 = f0 + 16;
      qi = f1 / (unsigned)(SEQ + 1);
      jr = f1 - qi * (unsigned)(SEQ + 1);
      float bd1 = jr ? bdraw[(size_t)qi * SEQ + jr - 1] : 0.f;
      float sv0 = s0[v] + bd0;
      float sv1 = s1[v] + bd1;
      float t = fmaxf(sv0, sv1);
#pragma unroll
      for (int off = 8; off >= 1; off >>= 1) t = fmaxf(t, __shfl_xor(t, off, 16));
      float nm = fmaxf(mrun[v], t);
      float sc = __expf(mrun[v] - nm);
      float e0 = __expf(sv0 - nm);
      float e1 = __expf(sv1 - nm);
      float rs = e0 + e1;
#pragma unroll
      for (int off = 8; off >= 1; off >>= 1) rs += __shfl_xor(rs, off, 16);
      lrun[v] = lrun[v] * sc + rs;
      mrun[v] = nm;
      p0[v] = e0;
      p1[v] = e1;
      oacc[0][v] *= sc; oacc[1][v] *= sc; oacc[2][v] *= sc; oacc[3][v] *= sc;
    }
    // ---- C-layout P -> LDS -> A-fragment layout --------------------------
#pragma unroll
    for (int v = 0; v < 8; ++v) {
      pbuf[(v + 8 * hf) * PST + l16]      = (__bf16)p0[v];
      pbuf[(v + 8 * hf) * PST + 16 + l16] = (__bf16)p1[v];
    }
    __syncthreads();  // uniform trip count; cross-lane LDS visibility
    v16bf ap;
    {
      const __bf16* pr = pbuf + l16 * PST + hf * 8;
      v8bf lo = *(const v8bf*)pr;
      v8bf hi = *(const v8bf*)(pr + 16);
#pragma unroll
      for (int i = 0; i < 8; ++i) { ap[i] = lo[i]; ap[i + 8] = hi[i]; }
    }
    // ---- O += P @ V^T over 64 output dims (4 tiles) ----------------------
#pragma unroll
    for (int t = 0; t < 4; ++t) {
      v16bf bv = load_b_fragT(Vt, t * 16 + l16, SEQ, kt, lane);
      oacc[t] = wmma_bf16(ap, bv, oacc[t]);
    }
    __syncthreads();
  }
  // ---- normalize and scatter into attn_out[BS, DM] (bf16) ----------------
#pragma unroll
  for (int t = 0; t < 4; ++t)
#pragma unroll
    for (int v = 0; v < 8; ++v) {
      float val = oacc[t][v] / lrun[v];
      int m = b * SEQ + q0 + v + 8 * hf;
      attnout[(size_t)m * DM + h * DH + t * 16 + l16] = (__bf16)val;
    }
}

// ---------------------------------------------------------------------------
// GEMM2: y = attn_out @ w_fc + b_fc + x   (residual fused; fp32 out)
// ---------------------------------------------------------------------------
__global__ __launch_bounds__(256) void k_gemm_fc(const __bf16* __restrict__ ab,
                                                 const __bf16* __restrict__ wT,
                                                 const float* __restrict__ bias,
                                                 const float* __restrict__ xres,
                                                 float* __restrict__ ybuf) {
  int lane = threadIdx.x & 31, wave = threadIdx.x >> 5;
  int l16 = lane & 15, hf = lane >> 4;
  int m0 = blockIdx.y * 64 + (wave >> 2) * 32;
  int n0 = blockIdx.x * 128 + (wave & 3) * 32;
  v8f acc[2][2] = {};
  gemm_tile_staged(ab, wT, blockIdx.y * 64, blockIdx.x * 128, acc);
#pragma unroll
  for (int i = 0; i < 2; ++i)
#pragma unroll
    for (int j = 0; j < 2; ++j) {
      int n = n0 + 16 * j + l16;
      float bv = bias[n];
#pragma unroll
      for (int v = 0; v < 8; ++v) {
        int m = m0 + 16 * i + v + 8 * hf;
        ybuf[(size_t)m * DM + n] = acc[i][j][v] + bv + xres[(size_t)m * DM + n];
      }
    }
}

// ---------------------------------------------------------------------------
// Post-LN: one 256-thread block per row of [BS, DM].
// ---------------------------------------------------------------------------
__global__ __launch_bounds__(256) void k_ln(const float* __restrict__ y,
                                            const float* __restrict__ g,
                                            const float* __restrict__ be,
                                            float* __restrict__ out) {
  __shared__ float sbuf[16];
  __shared__ float smu, srstd;
  int m = blockIdx.x;
  const float* row = y + (size_t)m * DM;
  float s = 0.f, s2 = 0.f;
  for (int i = threadIdx.x; i < DM; i += 256) {
    float v = row[i];
    s += v;
    s2 = fmaf(v, v, s2);
  }
  int lane = threadIdx.x & 31, wv = threadIdx.x >> 5;
#pragma unroll
  for (int off = 16; off >= 1; off >>= 1) {
    s += __shfl_xor(s, off);
    s2 += __shfl_xor(s2, off);
  }
  if (lane == 0) { sbuf[wv] = s; sbuf[8 + wv] = s2; }
  __syncthreads();
  if (threadIdx.x == 0) {
    float ts = 0.f, ts2 = 0.f;
#pragma unroll
    for (int i = 0; i < 8; ++i) { ts += sbuf[i]; ts2 += sbuf[8 + i]; }
    float mu = ts / (float)DM;
    float var = ts2 / (float)DM - mu * mu;
    smu = mu;
    srstd = rsqrtf(var + 1e-5f);
  }
  __syncthreads();
  float mu = smu, rstd = srstd;
  for (int i = threadIdx.x; i < DM; i += 256)
    out[(size_t)m * DM + i] = (row[i] - mu) * rstd * g[i] + be[i];
}

// ---------------------------------------------------------------------------
// Host launcher
// ---------------------------------------------------------------------------
extern "C" void kernel_launch(void* const* d_in, const int* in_sizes, int n_in,
                              void* d_out, int out_size, void* d_ws, size_t ws_size,
                              hipStream_t stream) {
  (void)in_sizes; (void)n_in; (void)out_size; (void)ws_size;
  const float* x    = (const float*)d_in[0];
  const float* pos  = (const float*)d_in[1];
  const float* wqkv = (const float*)d_in[2];
  const float* bqkv = (const float*)d_in[3];
  const float* wfc  = (const float*)d_in[4];
  const float* bfc  = (const float*)d_in[5];
  const float* lng  = (const float*)d_in[6];
  const float* lnb  = (const float*)d_in[7];
  float* out = (float*)d_out;

  char* p = (char*)d_ws;
  auto carve = [&](size_t bytes) -> char* {
    char* r = p;
    p += (bytes + 255) & ~(size_t)255;
    return r;
  };
  __bf16* xb     = (__bf16*)carve((size_t)BS * DM * 2);   // x bf16 [BS,DM]
  __bf16* posb   = (__bf16*)carve((size_t)SEQ * DM * 2);  // pos bf16 [S,DM]
  __bf16* wqkvT  = (__bf16*)carve((size_t)N3 * DM * 2);   // [N3,DM]
  __bf16* wfcT   = (__bf16*)carve((size_t)DM * DM * 2);   // [DM,DM]
  __bf16* qb     = (__bf16*)carve((size_t)BS * DM * 2);   // [B,H,S,dh] scaled
  __bf16* kb     = (__bf16*)carve((size_t)BS * DM * 2);   // [B,H,S,dh]
  __bf16* vTb    = (__bf16*)carve((size_t)BS * DM * 2);   // [B,H,dh,S]
  float*  bdraw  = (float*)carve((size_t)SEQ * SEQ * 4);  // one (b,h), 16MB
  __bf16* attnA  = (__bf16*)carve((size_t)BS * DM * 2);   // attn out bf16
  float*  ybuf   = (float*)carve((size_t)BS * DM * 4);    // pre-LN fp32

  k_cvt<<<(BS * DM + 255) / 256, 256, 0, stream>>>(x, xb, BS * DM);
  k_cvt<<<(SEQ * DM + 255) / 256, 256, 0, stream>>>(pos, posb, SEQ * DM);
  k_cvtT<<<(DM * N3 + 255) / 256, 256, 0, stream>>>(wqkv, wqkvT, DM, N3);
  k_cvtT<<<(DM * DM + 255) / 256, 256, 0, stream>>>(wfc, wfcT, DM, DM);

  k_gemm_qkv<<<dim3(N3 / 128, BS / 64), 256, 0, stream>>>(xb, wqkvT, bqkv, qb, kb, vTb);

  for (int b = 0; b < BATCH; ++b)
    for (int h = 0; h < NH; ++h) {
      k_gemm_bd<<<dim3(SEQ / 128, SEQ / 64), 256, 0, stream>>>(qb, posb, bdraw, b, h);
      k_attn<<<dim3(SEQ / 128), 256, 0, stream>>>(qb, kb, vTb, bdraw, attnA, b, h);
    }

  k_gemm_fc<<<dim3(DM / 128, BS / 64), 256, 0, stream>>>(attnA, wfcT, bfc, x, ybuf);
  k_ln<<<BS, 256, 0, stream>>>(ybuf, lng, lnb, out);
}
